// YATE_Attention_34883724378543
// MI455X (gfx1250) — compile-verified, matrix operand order checked by reading, not run on hardware
//
#include <hip/hip_runtime.h>
#include <hip/hip_bf16.h>

typedef __bf16 bf16;
typedef __bf16 v16bf __attribute__((ext_vector_type(16)));
typedef float  v8f   __attribute__((ext_vector_type(8)));

union Frag16 { v16bf v; uint4 u[2]; };

__device__ __forceinline__ v8f wmma_bf16(v16bf a, v16bf b, v8f c) {
  // D = A(16x32 bf16) x B(32x16 bf16) + C(16x16 f32)
  return __builtin_amdgcn_wmma_f32_16x16x32_bf16(false, a, false, b, (short)0, c, false, false);
}

// monotonic float <-> uint mapping for atomicMax on floats
__device__ __forceinline__ unsigned ford(float f) {
  unsigned u = __float_as_uint(f);
  return (u & 0x80000000u) ? ~u : (u | 0x80000000u);
}
__device__ __forceinline__ float funord(unsigned u) {
  unsigned v = (u & 0x80000000u) ? (u & 0x7FFFFFFFu) : ~u;
  return __uint_as_float(v);
}

// B-fragment loader: pre-permuted contiguous 16 bf16 per lane per (ntile,kstep)
__device__ __forceinline__ v16bf load_wfrag(const bf16* __restrict__ wf, int nt, int ks, int lane) {
  const bf16* p = wf + ((((nt << 2) + ks) * 32 + lane) << 4);
  Frag16 f;
  const uint4* q = (const uint4*)p;
  f.u[0] = q[0];
  f.u[1] = q[1];
  return f.v;
}

// A-fragment from LDS row-major bf16 tile (row stride 128)
// lane<16: rows 0..15, K chunks [0..7] and [16..23]; lane>=16: same rows, K [8..15],[24..31]
__device__ __forceinline__ v16bf load_afrag(const bf16* __restrict__ Zrow, int ks, int lane) {
  const bf16* p = Zrow + ks * 32 + ((lane < 16) ? 0 : 8);
  Frag16 f;
  f.u[0] = *(const uint4*)p;
  f.u[1] = *(const uint4*)(p + 16);
  return f.v;
}

// ---------------- zero scratch (atomics accumulate; must clear each call) ---------
__global__ void k_zero(unsigned* __restrict__ p, int n) {
  int i = blockIdx.x * blockDim.x + threadIdx.x;
  int stride = gridDim.x * blockDim.x;
  for (; i < n; i += stride) p[i] = 0u;
}

// ---------------- permute weights into B-fragment order (f32 -> bf16) -------------
__global__ void k_permute(const float* __restrict__ W0, const float* __restrict__ W1,
                          const float* __restrict__ W2, const float* __restrict__ W3,
                          bf16* __restrict__ out) {
  int id = blockIdx.x * blockDim.x + threadIdx.x;   // 4 * 8 * 4 * 32 * 16 = 65536
  if (id >= 4 * 8 * 4 * 32 * 16) return;
  int j     = id & 15;
  int lane  = (id >> 4) & 31;
  int ks    = (id >> 9) & 3;
  int nt    = (id >> 11) & 7;
  int mat   = id >> 14;
  const float* W = (mat == 0) ? W0 : (mat == 1) ? W1 : (mat == 2) ? W2 : W3;
  int ncol = nt * 16 + (lane & 15);
  int half = (lane < 16) ? 0 : 8;
  int k    = ks * 32 + ((j < 8) ? (half + j) : (16 + half + (j - 8)));
  // B[k][n] = W[n][k]  (GEMM is  Z @ W^T)
  out[(size_t)mat * (8 * 4 * 32 * 16) + ((((nt << 2) + ks) * 32 + lane) << 4) + j] =
      (bf16)W[ncol * 128 + k];
}

// ---------------- q = x @ Wq^T + bq  (bf16 WMMA, f32 accumulate) ------------------
__global__ __launch_bounds__(128) void k_qgemm(const float* __restrict__ x,
                                               const float* __restrict__ bq,
                                               const bf16* __restrict__ wfq,
                                               float* __restrict__ qout, int N) {
  __shared__ __align__(16) bf16 Xs[64 * 128];
  int tid = threadIdx.x, lane = tid & 31, w = tid >> 5;
  int rowBase = blockIdx.x * 64;
  for (int it = 0; it < 16; ++it) {
    int idx = it * 128 + tid;
    int e = idx >> 5, c4 = idx & 31;
    int g = rowBase + e; if (g >= N) g = N - 1;
    float4 xv = ((const float4*)(x + (size_t)g * 128))[c4];
    bf16* d = &Xs[e * 128 + c4 * 4];
    d[0] = (bf16)xv.x; d[1] = (bf16)xv.y; d[2] = (bf16)xv.z; d[3] = (bf16)xv.w;
  }
  __syncthreads();
  int tileRow = rowBase + w * 16;
  if (tileRow >= N) return;   // wave-uniform; active waves keep EXEC all-ones
  const bf16* Zrow = &Xs[(w * 16 + (lane & 15)) * 128];
  v16bf a0 = load_afrag(Zrow, 0, lane);
  v16bf a1 = load_afrag(Zrow, 1, lane);
  v16bf a2 = load_afrag(Zrow, 2, lane);
  v16bf a3 = load_afrag(Zrow, 3, lane);
  int half = (lane < 16) ? 0 : 8;
  for (int nt = 0; nt < 8; ++nt) {
    v8f acc = {};
    acc = wmma_bf16(a0, load_wfrag(wfq, nt, 0, lane), acc);
    acc = wmma_bf16(a1, load_wfrag(wfq, nt, 1, lane), acc);
    acc = wmma_bf16(a2, load_wfrag(wfq, nt, 2, lane), acc);
    acc = wmma_bf16(a3, load_wfrag(wfq, nt, 3, lane), acc);
    int ncol = nt * 16 + (lane & 15);
    float b = bq[ncol];
    for (int r = 0; r < 8; ++r)
      qout[(size_t)(tileRow + r + half) * 128 + ncol] = acc[r] + b;
  }
}

// ---------------- Pass A: Z, k, scores + segment-max, edge_out --------------------
__global__ __launch_bounds__(128) void k_edgeA(
    const float* __restrict__ x, const int* __restrict__ ei, const float* __restrict__ ea,
    const float* __restrict__ bk, const float* __restrict__ be,
    const float* __restrict__ qbuf,
    const bf16* __restrict__ wfk, const bf16* __restrict__ wfe,
    float* __restrict__ scores, unsigned* __restrict__ mord,
    float* __restrict__ edge_out, int E) {
  __shared__ __align__(16) bf16  Zs[64 * 128];
  __shared__ __align__(16) float Qs[64 * 128];
  __shared__ int dsts[64];
  __shared__ int srcs[64];
  int tid = threadIdx.x, lane = tid & 31, w = tid >> 5;
  int eBase = blockIdx.x * 64;
  if (tid < 64) { srcs[tid] = ei[eBase + tid]; dsts[tid] = ei[E + eBase + tid]; }
  __syncthreads();
  for (int it = 0; it < 16; ++it) {
    int idx = it * 128 + tid;
    int e = idx >> 5, c4 = idx & 31;
    float4 av = ((const float4*)(ea + (size_t)(eBase + e) * 128))[c4];
    float4 xv = ((const float4*)(x + (size_t)dsts[e] * 128))[c4];
    bf16* d = &Zs[e * 128 + c4 * 4];
    d[0] = (bf16)(av.x * xv.x); d[1] = (bf16)(av.y * xv.y);
    d[2] = (bf16)(av.z * xv.z); d[3] = (bf16)(av.w * xv.w);
    ((float4*)(&Qs[e * 128]))[c4] = ((const float4*)(qbuf + (size_t)srcs[e] * 128))[c4];
  }
  __syncthreads();
  const bf16* Zrow = &Zs[(w * 16 + (lane & 15)) * 128];
  v16bf a0 = load_afrag(Zrow, 0, lane);
  v16bf a1 = load_afrag(Zrow, 1, lane);
  v16bf a2 = load_afrag(Zrow, 2, lane);
  v16bf a3 = load_afrag(Zrow, 3, lane);
  int half = (lane < 16) ? 0 : 8;
  for (int nt = 0; nt < 8; ++nt) {           // head h == nt (C == 16 == tile width)
    int ncol = nt * 16 + (lane & 15);
    // --- k tile ---
    v8f acck = {};
    acck = wmma_bf16(a0, load_wfrag(wfk, nt, 0, lane), acck);
    acck = wmma_bf16(a1, load_wfrag(wfk, nt, 1, lane), acck);
    acck = wmma_bf16(a2, load_wfrag(wfk, nt, 2, lane), acck);
    acck = wmma_bf16(a3, load_wfrag(wfk, nt, 3, lane), acck);
    float bkv = bk[ncol];
    for (int r = 0; r < 8; ++r) {
      int lrow = w * 16 + r + half;
      float p = Qs[lrow * 128 + ncol] * (acck[r] + bkv);
      p += __shfl_xor(p, 1, 32); p += __shfl_xor(p, 2, 32);
      p += __shfl_xor(p, 4, 32); p += __shfl_xor(p, 8, 32);
      if ((lane & 15) == 0) {
        float s = p * 0.25f;                 // 1/sqrt(16)
        scores[(size_t)(eBase + lrow) * 8 + nt] = s;
        atomicMax(&mord[(size_t)dsts[lrow] * 8 + nt], ford(s));
      }
    }
    // --- edge_out tile ---
    v8f acce = {};
    acce = wmma_bf16(a0, load_wfrag(wfe, nt, 0, lane), acce);
    acce = wmma_bf16(a1, load_wfrag(wfe, nt, 1, lane), acce);
    acce = wmma_bf16(a2, load_wfrag(wfe, nt, 2, lane), acce);
    acce = wmma_bf16(a3, load_wfrag(wfe, nt, 3, lane), acce);
    float bev = be[ncol];
    for (int r = 0; r < 8; ++r) {
      int lrow = w * 16 + r + half;
      edge_out[(size_t)(eBase + lrow) * 128 + ncol] = acce[r] + bev;
    }
  }
}

// ---------------- Pass B: recompute Z, v; ex = exp(s-m); scatter ------------------
__global__ __launch_bounds__(128) void k_edgeB(
    const float* __restrict__ x, const int* __restrict__ ei, const float* __restrict__ ea,
    const float* __restrict__ bv, const bf16* __restrict__ wfv,
    const float* __restrict__ scores, const unsigned* __restrict__ mord,
    float* __restrict__ denom, float* __restrict__ outraw, int E) {
  __shared__ __align__(16) bf16 Zs[64 * 128];
  __shared__ float Exs[64 * 8];
  __shared__ int dsts[64];
  int tid = threadIdx.x, lane = tid & 31, w = tid >> 5;
  int eBase = blockIdx.x * 64;
  if (tid < 64) dsts[tid] = ei[E + eBase + tid];
  __syncthreads();
  for (int it = 0; it < 16; ++it) {
    int idx = it * 128 + tid;
    int e = idx >> 5, c4 = idx & 31;
    float4 av = ((const float4*)(ea + (size_t)(eBase + e) * 128))[c4];
    float4 xv = ((const float4*)(x + (size_t)dsts[e] * 128))[c4];
    bf16* d = &Zs[e * 128 + c4 * 4];
    d[0] = (bf16)(av.x * xv.x); d[1] = (bf16)(av.y * xv.y);
    d[2] = (bf16)(av.z * xv.z); d[3] = (bf16)(av.w * xv.w);
  }
  for (int i = tid; i < 512; i += 128) {
    int e = i >> 3, h = i & 7;
    float sc = scores[(size_t)(eBase + e) * 8 + h];
    float mv = funord(mord[(size_t)dsts[e] * 8 + h]);
    float ex = __expf(sc - mv);
    Exs[e * 8 + h] = ex;
    atomicAdd(&denom[(size_t)dsts[e] * 8 + h], ex);
  }
  __syncthreads();
  const bf16* Zrow = &Zs[(w * 16 + (lane & 15)) * 128];
  v16bf a0 = load_afrag(Zrow, 0, lane);
  v16bf a1 = load_afrag(Zrow, 1, lane);
  v16bf a2 = load_afrag(Zrow, 2, lane);
  v16bf a3 = load_afrag(Zrow, 3, lane);
  int half = (lane < 16) ? 0 : 8;
  for (int nt = 0; nt < 8; ++nt) {
    v8f acc = {};
    acc = wmma_bf16(a0, load_wfrag(wfv, nt, 0, lane), acc);
    acc = wmma_bf16(a1, load_wfrag(wfv, nt, 1, lane), acc);
    acc = wmma_bf16(a2, load_wfrag(wfv, nt, 2, lane), acc);
    acc = wmma_bf16(a3, load_wfrag(wfv, nt, 3, lane), acc);
    int ncol = nt * 16 + (lane & 15);
    float bvv = bv[ncol];
    for (int r = 0; r < 8; ++r) {
      int lrow = w * 16 + r + half;
      float val = (acc[r] + bvv) * Exs[lrow * 8 + nt];
      atomicAdd(&outraw[(size_t)dsts[lrow] * 128 + ncol], val);
    }
  }
}

// ---------------- finalize: out = out_raw / denom ---------------------------------
__global__ void k_final(const float* __restrict__ raw, const float* __restrict__ denom,
                        float* __restrict__ out, int total) {
  int i = blockIdx.x * blockDim.x + threadIdx.x;
  if (i >= total) return;
  int n = i >> 7, d = i & 127, h = d >> 4;
  float den = denom[n * 8 + h];
  out[i] = (den > 0.f) ? raw[i] / den : 0.f;
}

extern "C" void kernel_launch(void* const* d_in, const int* in_sizes, int n_in,
                              void* d_out, int out_size, void* d_ws, size_t ws_size,
                              hipStream_t stream) {
  const float* x  = (const float*)d_in[0];
  const int*   ei = (const int*)d_in[1];    // [2,E]: row0 src, row1 dst
  const float* ea = (const float*)d_in[2];
  const float* Wq = (const float*)d_in[3];
  const float* bq = (const float*)d_in[4];
  const float* Wk = (const float*)d_in[5];
  const float* bk = (const float*)d_in[6];
  const float* Wv = (const float*)d_in[7];
  const float* bv = (const float*)d_in[8];
  const float* We = (const float*)d_in[9];
  const float* be = (const float*)d_in[10];

  const int D = 128;
  const int N = in_sizes[0] / D;            // 50000 (multiple of 16)
  const int E = in_sizes[2] / D;            // 800000 (multiple of 64)

  char* ws = (char*)d_ws;
  size_t off = 0;
  float* qbuf = (float*)(ws + off);   off += (size_t)N * D * 4;
  float* scores = (float*)(ws + off); off += (size_t)E * 8 * 4;
  size_t zero_off = off;
  unsigned* mord = (unsigned*)(ws + off); off += (size_t)N * 8 * 4;
  float* denom = (float*)(ws + off);      off += (size_t)N * 8 * 4;
  float* outraw = (float*)(ws + off);     off += (size_t)N * D * 4;
  bf16* wfrag = (bf16*)(ws + off);        off += (size_t)4 * 8 * 4 * 32 * 16 * 2;
  const int WFSZ = 8 * 4 * 32 * 16;
  bf16* wfq = wfrag + 0 * WFSZ;
  bf16* wfk = wfrag + 1 * WFSZ;
  bf16* wfv = wfrag + 2 * WFSZ;
  bf16* wfe = wfrag + 3 * WFSZ;

  float* out_nodes = (float*)d_out;
  float* edge_out  = (float*)d_out + (size_t)N * D;

  int zero_cnt = N * 8 + N * 8 + N * D;   // mord + denom + outraw (contiguous)
  k_zero<<<2048, 256, 0, stream>>>((unsigned*)(ws + zero_off), zero_cnt);
  k_permute<<<(4 * WFSZ + 255) / 256, 256, 0, stream>>>(Wq, Wk, Wv, We, wfrag);
  k_qgemm<<<(N + 63) / 64, 128, 0, stream>>>(x, bq, wfq, qbuf, N);
  k_edgeA<<<E / 64, 128, 0, stream>>>(x, ei, ea, bk, be, qbuf, wfk, wfe,
                                      scores, mord, edge_out, E);
  k_edgeB<<<E / 64, 128, 0, stream>>>(x, ei, ea, bv, wfv, scores, mord,
                                      denom, outraw, E);
  k_final<<<((size_t)N * D + 255) / 256, 256, 0, stream>>>(outraw, denom, out_nodes, N * D);
}
